// NC_35673998361024
// MI455X (gfx1250) — matrix operand order repaired and checked by reading, not compiled, not run on hardware
//
#include <hip/hip_runtime.h>
#include <hip/hip_bf16.h>
#include <math.h>

// ---------------------------------------------------------------------------
// MI455X (gfx1250, wave32). All matmuls (corr einsum, 4D-conv implicit GEMMs,
// attention @ V) use v_wmma_f32_16x16x32_bf16 with f32 accumulation.
// Conv stage stages its (3x3 ij-neighborhood x Cin) input tile + weights into
// LDS via global_load_async_to_lds_b128 (ASYNCcnt) -- the CDNA5 async path.
// ---------------------------------------------------------------------------

typedef __attribute__((ext_vector_type(16))) __bf16          v16bf;
typedef __attribute__((ext_vector_type(8)))  float           v8f;
typedef __attribute__((ext_vector_type(8)))  unsigned short  v8u16;

union BFrag {
    v16bf bf;
    struct Pair { v8u16 lo, hi; } p;
    unsigned short us[16];
};

__device__ __forceinline__ unsigned short f2bf(float f) {
    unsigned int u = __builtin_bit_cast(unsigned int, f);
    unsigned int r = (u + 0x7FFFu + ((u >> 16) & 1u)) >> 16;   // RNE
    return (unsigned short)r;
}

__device__ __forceinline__ float wave_sum(float v) {
    #pragma unroll
    for (int off = 16; off >= 1; off >>= 1) v += __shfl_xor(v, off, 32);
    return v;
}
__device__ __forceinline__ float wave_max(float v) {
    #pragma unroll
    for (int off = 16; off >= 1; off >>= 1) v = fmaxf(v, __shfl_xor(v, off, 32));
    return v;
}

// Async global->LDS copy of 16 bytes (per active lane). Tracked by ASYNCcnt.
__device__ __forceinline__ void async_b128_to_lds(unsigned lds_off,
                                                  unsigned long long gaddr) {
    asm volatile("global_load_async_to_lds_b128 %0, %1, off"
                 :: "v"(lds_off), "v"(gaddr) : "memory");
}
__device__ __forceinline__ void wait_async0() {
    asm volatile("s_wait_asynccnt 0" ::: "memory");
}
__device__ __forceinline__ unsigned lds_off32(const void* p) {
    // addrspace(3) -> generic keeps the LDS byte offset in the low 32 bits
    return (unsigned)(unsigned long long)p;
}

// ---------------------------------------------------------------------------
// K1: channel l2-norm of [L,B,1024,400] + transpose/pack to bf16 [LB][400][1024]
// One wave per (lb, ij). 18*400 = 7200 waves.
// ---------------------------------------------------------------------------
__global__ void nrm_pack_kernel(const float* __restrict__ in,
                                unsigned short* __restrict__ out) {
    int wave = (blockIdx.x * blockDim.x + threadIdx.x) >> 5;
    int lane = threadIdx.x & 31;
    if (wave >= 18 * 400) return;
    int lb = wave / 400, ij = wave % 400;
    const float* src = in + (size_t)lb * 1024 * 400;
    float v[32];
    float s = 0.f;
    #pragma unroll
    for (int k = 0; k < 32; ++k) {
        float x = src[(size_t)(k * 32 + lane) * 400 + ij];
        v[k] = x;
        s += x * x;
    }
    s = wave_sum(s);
    float inv = 1.0f / fmaxf(sqrtf(s), 1e-12f);
    unsigned short* dst = out + (size_t)lb * 400 * 1024 + (size_t)ij * 1024;
    #pragma unroll
    for (int k = 0; k < 32; ++k) dst[k * 32 + lane] = f2bf(v[k] * inv);
}

// ---------------------------------------------------------------------------
// K2: corr[b][l][ij][km] = sum_c Aq[ij,c]*As[km,c]  (M=N=400, K=1024) x18.
// One wave per 16x16 tile, 18*25*25 = 11250 tiles. Writes corr and corr^T bf16.
// ---------------------------------------------------------------------------
__global__ void corr_gemm_kernel(const unsigned short* __restrict__ Aq,
                                 const unsigned short* __restrict__ As,
                                 unsigned short* __restrict__ corr,
                                 unsigned short* __restrict__ corrT) {
    int wave = (blockIdx.x * blockDim.x + threadIdx.x) >> 5;
    int lane = threadIdx.x & 31;
    if (wave >= 18 * 625) return;
    int lb = wave / 625;
    int r  = wave % 625;
    int mt = r / 25, nt = r % 25;
    int b = lb % 2, l = lb / 2;
    int row = lane & 15, hi = lane >> 4;

    const unsigned short* ap = Aq + (size_t)lb * 400 * 1024 + (size_t)(mt * 16 + row) * 1024 + hi * 8;
    const unsigned short* bp = As + (size_t)lb * 400 * 1024 + (size_t)(nt * 16 + row) * 1024 + hi * 8;

    v8f acc = {};
    #pragma unroll 4
    for (int k0 = 0; k0 < 1024; k0 += 32) {
        BFrag a, w;
        a.p.lo = *(const v8u16*)(ap + k0);
        a.p.hi = *(const v8u16*)(ap + k0 + 16);
        w.p.lo = *(const v8u16*)(bp + k0);
        w.p.hi = *(const v8u16*)(bp + k0 + 16);
        acc = __builtin_amdgcn_wmma_f32_16x16x32_bf16(false, a.bf, false, w.bf,
                                                      (short)0, acc, false, false);
    }
    size_t cbase = ((size_t)b * 9 + l) * 400;
    int ij0 = mt * 16, km0 = nt * 16, n = lane & 15;
    #pragma unroll
    for (int rr = 0; rr < 8; ++rr) {
        int mm = rr + hi * 8;
        unsigned short q = f2bf(acc[rr]);
        corr [(cbase + ij0 + mm) * 400 + km0 + n ] = q;
        corrT[(cbase + km0 + n ) * 400 + ij0 + mm] = q;
    }
}

// ---------------------------------------------------------------------------
// K3a: pack conv weights [Cout][Cin][3,3,3,3] f32 -> bf16 GEMM-B [16][Kpad]
// ---------------------------------------------------------------------------
__global__ void pack_w_kernel(const float* __restrict__ w,
                              unsigned short* __restrict__ out,
                              int Cout, int Cin, int Kpad) {
    int idx = blockIdx.x * blockDim.x + threadIdx.x;
    if (idx >= 16 * Kpad) return;
    int co = idx / Kpad, kk = idx % Kpad;
    float v = 0.f;
    if (co < Cout && kk < Cin * 81) {
        int ci = kk / 81, t = kk % 81;
        v = w[((size_t)co * Cin + ci) * 81 + t];
    }
    out[idx] = f2bf(v);
}

// ---------------------------------------------------------------------------
// K3b: 4D conv as implicit GEMM, LDS-staged.
// One block per (b, ij): async-stage 3x3 ij-neighborhood x Cin km-planes
// (zero rows for out-of-grid neighbors) + packed weights into LDS, then
// 8 waves cover the 25 km-tiles. K = Cin*81; kk/9 indexes the staged row.
// ---------------------------------------------------------------------------
__device__ __forceinline__ unsigned short lds_gather(
        const unsigned short* __restrict__ sx, int Ktot, int k, int m, int kk) {
    if (kk >= Ktot) return 0;
    int row = kk / 9;            // ci*9 + (di*3+dj)
    int t2  = kk - row * 9;      // dk*3+dm
    int dk = t2 / 3, dm = t2 - dk * 3;
    int k2 = k + dk - 1, mm = m + dm - 1;
    if ((unsigned)k2 >= 20u || (unsigned)mm >= 20u) return 0;
    return sx[row * 400 + k2 * 20 + mm];
}

__global__ void conv4d_wmma_kernel(const unsigned short* __restrict__ X, int Ci,
                                   const unsigned short* __restrict__ W2d,
                                   int Ktot, int Kpad,
                                   unsigned short* __restrict__ Ybf,
                                   float* __restrict__ Yf32,
                                   int CoStore) {
    extern __shared__ unsigned short smem[];
    unsigned short* sx = smem;                  // [Ci*9][400] input rows
    unsigned short* sw = smem + Ci * 9 * 400;   // [16][Kpad]  weights

    int tid = threadIdx.x;
    int bij = blockIdx.x;                 // 0..799
    int b = bij / 400, ij = bij % 400;
    int i = ij / 20, j = ij % 20;

    // ---- stage input neighborhood (async DMA; zero-pad OOB ij rows) ----
    int xchunks = Ci * 9 * 50;            // 16B chunks of input rows
    v8u16 zz = {};
    for (int c = tid; c < xchunks; c += 256) {
        int row = c / 50, piece = c % 50;
        int ci = row / 9, t = row % 9;
        int ii = i + t / 3 - 1, jj = j + t % 3 - 1;
        unsigned short* dst = sx + row * 400 + piece * 8;
        if ((unsigned)ii < 20u && (unsigned)jj < 20u) {
            unsigned long long g = (unsigned long long)
                (X + ((((size_t)b * Ci + ci) * 400) + ii * 20 + jj) * 400 + piece * 8);
            async_b128_to_lds(lds_off32(dst), g);
        } else {
            *(v8u16*)dst = zz;            // DS store; disjoint from async dests
        }
    }
    // ---- stage weights ----
    int wchunks = 16 * (Kpad / 8);
    for (int c = tid; c < wchunks; c += 256) {
        async_b128_to_lds(lds_off32(sw + c * 8),
                          (unsigned long long)(W2d + c * 8));
    }
    wait_async0();
    __syncthreads();

    // ---- compute: 25 km-tiles spread over 8 waves ----
    int lane = tid & 31, wv = tid >> 5;
    int rowl = lane & 15, hi = lane >> 4;
    int kbA = hi * 8;
    int ksteps = Kpad / 32;
    for (int tile = wv; tile < 25; tile += 8) {
        int km = tile * 16 + rowl;
        int k = km / 20, m = km % 20;
        v8f acc = {};
        for (int ks = 0; ks < ksteps; ++ks) {
            int k0 = ks * 32;
            BFrag a;
            #pragma unroll
            for (int e = 0; e < 8; ++e) {
                a.us[e]     = lds_gather(sx, Ktot, k, m, k0 + kbA + e);
                a.us[e + 8] = lds_gather(sx, Ktot, k, m, k0 + kbA + 16 + e);
            }
            BFrag w;
            const unsigned short* wp = sw + (size_t)rowl * Kpad + k0 + hi * 8;
            w.p.lo = *(const v8u16*)wp;
            w.p.hi = *(const v8u16*)(wp + 16);
            acc = __builtin_amdgcn_wmma_f32_16x16x32_bf16(false, a.bf, false, w.bf,
                                                          (short)0, acc, false, false);
        }
        int n = lane & 15;
        if (Yf32) {                       // last layer: Cout=1, f32 + relu
            if (n == 0) {
                #pragma unroll
                for (int rr = 0; rr < 8; ++rr) {
                    int kmo = tile * 16 + rr + hi * 8;
                    Yf32[((size_t)b * 400 + ij) * 400 + kmo] = fmaxf(acc[rr], 0.f);
                }
            }
        } else {                          // hidden layers: bf16 + relu
            if (n < CoStore) {
                #pragma unroll
                for (int rr = 0; rr < 8; ++rr) {
                    int kmo = tile * 16 + rr + hi * 8;
                    Ybf[(((size_t)b * CoStore + n) * 400 + ij) * 400 + kmo] =
                        f2bf(fmaxf(acc[rr], 0.f));
                }
            }
        }
    }
}

// ---------------------------------------------------------------------------
// K4a: softmax over s of 20*(yA[b,q,s] + yB[b,s,q]) -> attn bf16 [b][q][416]
// ---------------------------------------------------------------------------
__global__ void softmax_kernel(const float* __restrict__ yA,
                               const float* __restrict__ yB,
                               unsigned short* __restrict__ attn) {
    int wave = (blockIdx.x * blockDim.x + threadIdx.x) >> 5;
    int lane = threadIdx.x & 31;
    if (wave >= 800) return;
    int b = wave / 400, q = wave % 400;
    const float* rA = yA + ((size_t)b * 400 + q) * 400;
    const float* cB = yB + (size_t)b * 400 * 400 + q;

    float mx = -1e30f;
    for (int s = lane; s < 400; s += 32)
        mx = fmaxf(mx, 20.f * (rA[s] + cB[(size_t)s * 400]));
    mx = wave_max(mx);

    float sum = 0.f;
    for (int s = lane; s < 400; s += 32)
        sum += __expf(20.f * (rA[s] + cB[(size_t)s * 400]) - mx);
    sum = wave_sum(sum);
    float rs = 1.0f / sum;

    unsigned short* arow = attn + ((size_t)b * 400 + q) * 416;
    for (int s = lane; s < 416; s += 32) {
        float p = 0.f;
        if (s < 400)
            p = __expf(20.f * (rA[s] + cB[(size_t)s * 400]) - mx) * rs;
        arow[s] = f2bf(p);
    }
}

// K4a': pack f_s [b][c][400] f32 -> vbf [b][c][416] bf16 (zero-padded K).
__global__ void pack_v_kernel(const float* __restrict__ fs,
                              unsigned short* __restrict__ vbf) {
    int idx = blockIdx.x * blockDim.x + threadIdx.x;
    if (idx >= 2 * 256 * 416) return;
    int s = idx % 416, bc = idx / 416;
    float v = (s < 400) ? fs[(size_t)bc * 400 + s] : 0.f;
    vbf[idx] = f2bf(v);
}

// ---------------------------------------------------------------------------
// K4b: att_fq[b,c,q] = sum_s attn[b,q,s]*v[b,c,s].  M=400(q), N=256(c), K=416.
// ---------------------------------------------------------------------------
__global__ void attn_gemm_kernel(const unsigned short* __restrict__ attn,
                                 const unsigned short* __restrict__ vbf,
                                 float* __restrict__ att_out) {
    int wave = (blockIdx.x * blockDim.x + threadIdx.x) >> 5;
    int lane = threadIdx.x & 31;
    if (wave >= 800) return;
    int b  = wave / 400;
    int r  = wave % 400;
    int qt = r / 16, ct = r % 16;
    int row = lane & 15, hi = lane >> 4;

    const unsigned short* ap = attn + ((size_t)b * 400 + qt * 16 + row) * 416 + hi * 8;
    const unsigned short* vp = vbf  + ((size_t)b * 256 + ct * 16 + row) * 416 + hi * 8;

    v8f acc = {};
    #pragma unroll
    for (int k0 = 0; k0 < 416; k0 += 32) {
        BFrag a, w;
        a.p.lo = *(const v8u16*)(ap + k0);
        a.p.hi = *(const v8u16*)(ap + k0 + 16);
        w.p.lo = *(const v8u16*)(vp + k0);
        w.p.hi = *(const v8u16*)(vp + k0 + 16);
        acc = __builtin_amdgcn_wmma_f32_16x16x32_bf16(false, a.bf, false, w.bf,
                                                      (short)0, acc, false, false);
    }
    int n = lane & 15;
    #pragma unroll
    for (int rr = 0; rr < 8; ++rr) {
        int q = qt * 16 + rr + hi * 8;
        int c = ct * 16 + n;
        att_out[((size_t)b * 256 + c) * 400 + q] = acc[rr];
    }
}

// ---------------------------------------------------------------------------
// K5: fq = l2norm_c(f_q) + 0.5*l2norm_c(att_fq). One wave per (b,q).
// ---------------------------------------------------------------------------
__global__ void finalize_kernel(const float* __restrict__ fq,
                                const float* __restrict__ att,
                                float* __restrict__ out_fq) {
    int wave = (blockIdx.x * blockDim.x + threadIdx.x) >> 5;
    int lane = threadIdx.x & 31;
    if (wave >= 800) return;
    int b = wave / 400, q = wave % 400;
    const float* pf = fq  + (size_t)b * 256 * 400 + q;
    const float* pa = att + (size_t)b * 256 * 400 + q;
    float vf[8], va[8], s1 = 0.f, s2 = 0.f;
    #pragma unroll
    for (int k = 0; k < 8; ++k) {
        float a = pf[(size_t)(k * 32 + lane) * 400];
        float c = pa[(size_t)(k * 32 + lane) * 400];
        vf[k] = a; va[k] = c; s1 += a * a; s2 += c * c;
    }
    s1 = wave_sum(s1);
    s2 = wave_sum(s2);
    float i1 = 1.f / fmaxf(sqrtf(s1), 1e-12f);
    float i2 = 1.f / fmaxf(sqrtf(s2), 1e-12f);
    float* po = out_fq + (size_t)b * 256 * 400 + q;
    #pragma unroll
    for (int k = 0; k < 8; ++k)
        po[(size_t)(k * 32 + lane) * 400] = vf[k] * i1 + va[k] * i2 * 0.5f;
}

// ---------------------------------------------------------------------------
extern "C" void kernel_launch(void* const* d_in, const int* in_sizes, int n_in,
                              void* d_out, int out_size, void* d_ws, size_t ws_size,
                              hipStream_t stream) {
    (void)in_sizes; (void)n_in; (void)out_size; (void)ws_size;
    const float* fq_feats = (const float*)d_in[0];
    const float* fs_feats = (const float*)d_in[1];
    const float* f_q      = (const float*)d_in[2];
    const float* f_s      = (const float*)d_in[3];
    const float* w1       = (const float*)d_in[4];
    const float* w2       = (const float*)d_in[5];
    const float* w3       = (const float*)d_in[6];

    float* out_fq  = (float*)d_out;                 // [2][256][400]
    float* out_att = (float*)d_out + 2 * 256 * 400; // [2][256][400]

    auto align = [](size_t x) { return (x + 255) & ~(size_t)255; };
    char* ws = (char*)d_ws;
    size_t off = 0;
    auto carve = [&](size_t bytes) { void* p = ws + off; off += align(bytes); return p; };

    unsigned short* fqn   = (unsigned short*)carve((size_t)18 * 400 * 1024 * 2);
    unsigned short* fsn   = (unsigned short*)carve((size_t)18 * 400 * 1024 * 2);
    unsigned short* corr  = (unsigned short*)carve((size_t)18 * 400 * 400 * 2);
    unsigned short* corrT = (unsigned short*)carve((size_t)18 * 400 * 400 * 2);
    unsigned short* W1p   = (unsigned short*)carve((size_t)16 * 736 * 2);
    unsigned short* W2p   = (unsigned short*)carve((size_t)16 * 832 * 2);
    unsigned short* W3p   = (unsigned short*)carve((size_t)16 * 832 * 2);
    unsigned short* actA0 = (unsigned short*)carve((size_t)2 * 10 * 400 * 400 * 2);
    unsigned short* actA1 = (unsigned short*)carve((size_t)2 * 10 * 400 * 400 * 2);
    unsigned short* actB0 = (unsigned short*)carve((size_t)2 * 10 * 400 * 400 * 2);
    unsigned short* actB1 = (unsigned short*)carve((size_t)2 * 10 * 400 * 400 * 2);
    float*          y3A   = (float*)carve((size_t)2 * 400 * 400 * 4);
    float*          y3B   = (float*)carve((size_t)2 * 400 * 400 * 4);
    unsigned short* attn  = (unsigned short*)carve((size_t)2 * 400 * 416 * 2);
    unsigned short* vbf   = (unsigned short*)carve((size_t)2 * 256 * 416 * 2);

    const int BT = 256;   // 8 waves per block

    nrm_pack_kernel<<<900, BT, 0, stream>>>(fq_feats, fqn);
    nrm_pack_kernel<<<900, BT, 0, stream>>>(fs_feats, fsn);

    pack_w_kernel<<<(16 * 736 + BT - 1) / BT, BT, 0, stream>>>(w1, W1p, 10,  9, 736);
    pack_w_kernel<<<(16 * 832 + BT - 1) / BT, BT, 0, stream>>>(w2, W2p, 10, 10, 832);
    pack_w_kernel<<<(16 * 832 + BT - 1) / BT, BT, 0, stream>>>(w3, W3p,  1, 10, 832);
    pack_v_kernel<<<(2 * 256 * 416 + BT - 1) / BT, BT, 0, stream>>>(f_s, vbf);

    corr_gemm_kernel<<<1407, BT, 0, stream>>>(fqn, fsn, corr, corrT);

    // Conv stack: one block per (b,ij); dynamic LDS = input rows + weights.
    size_t sh1 = (size_t)(9  * 9 * 400 + 16 * 736) * 2;   // 88,352 B
    size_t sh2 = (size_t)(10 * 9 * 400 + 16 * 832) * 2;   // 98,624 B
    conv4d_wmma_kernel<<<800, BT, sh1, stream>>>(corr,  9, W1p, 729, 736, actA0, nullptr, 10);
    conv4d_wmma_kernel<<<800, BT, sh1, stream>>>(corrT, 9, W1p, 729, 736, actB0, nullptr, 10);
    conv4d_wmma_kernel<<<800, BT, sh2, stream>>>(actA0, 10, W2p, 810, 832, actA1, nullptr, 10);
    conv4d_wmma_kernel<<<800, BT, sh2, stream>>>(actB0, 10, W2p, 810, 832, actB1, nullptr, 10);
    conv4d_wmma_kernel<<<800, BT, sh2, stream>>>(actA1, 10, W3p, 810, 832, nullptr, y3A, 1);
    conv4d_wmma_kernel<<<800, BT, sh2, stream>>>(actB1, 10, W3p, 810, 832, nullptr, y3B, 1);

    softmax_kernel<<<100, BT, 0, stream>>>(y3A, y3B, attn);
    attn_gemm_kernel<<<100, BT, 0, stream>>>(attn, vbf, out_att);

    finalize_kernel<<<100, BT, 0, stream>>>(f_q, out_att, out_fq);
}
